// CPCModule_63565515981073
// MI455X (gfx1250) — matrix operand order repaired and checked by reading, not compiled, not run on hardware
//
#include <hip/hip_runtime.h>
#include <hip/hip_bf16.h>
#include <math.h>

typedef __attribute__((ext_vector_type(16))) _Float16 v16h;
typedef __attribute__((ext_vector_type(8)))  _Float16 h8;
typedef __attribute__((ext_vector_type(8)))  float    v8f;

#define BB   64
#define TT   512
#define FF   256
#define ENC  256
#define PP   128
#define HH   256
#define KP   12
#define INV_TEMP 10.0f

// ---------------------------------------------------------------------------
// WMMA fragment loaders (wave32, v_wmma_f32_16x16x32_f16 layouts)
// A 16x32 f16: lane m=lane&15, hi=lane>>4; halves 0..7 -> K=k0+hi*8+j,
//              halves 8..15 -> K=k0+16+hi*8+j   (two contiguous 8-half runs)
// B 32x16 f16: lane n=lane&15, hi=lane>>4; same K mapping as A (two 8-half runs)
//              (B stored transposed / N-major)
// C/D 16x16 f32: vgpr r, lane: M=r+hi*8, N=lane&15
// ---------------------------------------------------------------------------
__device__ __forceinline__ v16h load_a_h(const _Float16* A, int lda, int k0, int lane) {
  int m = lane & 15, hi = lane >> 4;
  const _Float16* row = A + (size_t)m * lda + k0 + hi * 8;
  h8 lo = *(const h8*)row;          // global_load_b128 / ds_load_b128
  h8 hv = *(const h8*)(row + 16);
  v16h a;
#pragma unroll
  for (int j = 0; j < 8; ++j) { a[j] = lo[j]; a[8 + j] = hv[j]; }
  return a;
}

// BT is the transposed (N-major, N x K) f16 matrix, ldk = K stride
__device__ __forceinline__ v16h load_b_t(const _Float16* BT, int ldk, int n0, int k0, int lane) {
  int n = lane & 15, hi = lane >> 4;
  const _Float16* p = BT + (size_t)(n0 + n) * ldk + k0 + hi * 8;
  h8 lo = *(const h8*)p;
  h8 hv = *(const h8*)(p + 16);
  v16h b;
#pragma unroll
  for (int j = 0; j < 8; ++j) { b[j] = lo[j]; b[8 + j] = hv[j]; }
  return b;
}

// same, with row clamp (for z_pos rows near the end of the sequence)
__device__ __forceinline__ v16h load_b_t_clamp(const _Float16* BT, int ldk, int n0, int k0,
                                               int maxrow, int lane) {
  int n = lane & 15, hi = lane >> 4;
  int row = n0 + n; if (row > maxrow) row = maxrow;
  const _Float16* p = BT + (size_t)row * ldk + k0 + hi * 8;
  h8 lo = *(const h8*)p;
  h8 hv = *(const h8*)(p + 16);
  v16h b;
#pragma unroll
  for (int j = 0; j < 8; ++j) { b[j] = lo[j]; b[8 + j] = hv[j]; }
  return b;
}

__device__ __forceinline__ v8f wmma_step(v16h a, v16h b, v8f c) {
  return __builtin_amdgcn_wmma_f32_16x16x32_f16(false, a, false, b, (short)0, c, false, false);
}

// ---------------------------------------------------------------------------
// Prep kernels: one-time f32 -> f16 convert and transpose-convert (tiny data)
// ---------------------------------------------------------------------------
__global__ void cvt_kernel(const float* __restrict__ src, _Float16* __restrict__ dst, int n) {
  for (int i = blockIdx.x * blockDim.x + threadIdx.x; i < n; i += gridDim.x * blockDim.x)
    dst[i] = (_Float16)src[i];
}

// dst[C x R] = transpose(src[R x C]), batched via blockIdx.y
__global__ void tr_kernel(const float* __restrict__ src, _Float16* __restrict__ dst,
                          int R, int C, int sb, int db) {
  src += (size_t)blockIdx.y * sb;
  dst += (size_t)blockIdx.y * db;
  int i = blockIdx.x * 256 + threadIdx.x;
  if (i < R * C) {
    int r = i / C, cc = i % C;
    dst[(size_t)cc * R + r] = (_Float16)src[i];
  }
}

// ---------------------------------------------------------------------------
// Kernel 1: fused encoder  z = (x@W_enc + b_enc) @ W_proj + b_proj  (f16 out)
// x tile staged into LDS with gfx1250 async-to-LDS copies (ASYNCcnt tracked).
// ---------------------------------------------------------------------------
__global__ void enc_kernel(const _Float16* __restrict__ xh, const _Float16* __restrict__ WencT,
                           const float* __restrict__ benc, const _Float16* __restrict__ WprojT,
                           const float* __restrict__ bproj, _Float16* __restrict__ z) {
  __shared__ __align__(16) _Float16 xs[16][FF];   // 8 KB
  __shared__ __align__(16) _Float16 ts[16][ENC];  // 8 KB
  const int tid = threadIdx.x;
  const int wave = tid >> 5, lane = tid & 31;
  const int m0 = blockIdx.x * 16;  // row block in [0, B*T)

  // async-stage the contiguous 8KB f16 x tile: 512 x b128 chunks, 4 per thread
  {
    unsigned lds_base = (unsigned)(uintptr_t)&xs[0][0];  // generic LDS ptr: low 32b = LDS addr
    const _Float16* gsrc = xh + (size_t)m0 * FF;
#pragma unroll
    for (int i = 0; i < 4; ++i) {
      unsigned idx = tid + i * 128;
      unsigned laddr = lds_base + idx * 16;
      unsigned long long gaddr = (unsigned long long)(uintptr_t)gsrc + (unsigned long long)idx * 16;
      asm volatile("global_load_async_to_lds_b128 %0, %1, off"
                   :: "v"(laddr), "v"(gaddr) : "memory");
    }
    asm volatile("s_wait_asynccnt 0" ::: "memory");
  }
  __syncthreads();

  const int nl = lane & 15, hi = lane >> 4;

  // stage 1: tmp(16x256) = xs @ Wenc + benc  (16 N-subtiles, 4 per wave)
  for (int s = wave; s < ENC / 16; s += 4) {
    int n0 = s * 16;
    v8f acc = {};
#pragma unroll
    for (int k0 = 0; k0 < FF; k0 += 32) {
      v16h a = load_a_h(&xs[0][0], FF, k0, lane);
      v16h b = load_b_t(WencT, FF, n0, k0, lane);
      acc = wmma_step(a, b, acc);
    }
    float bv = benc[n0 + nl];
#pragma unroll
    for (int r = 0; r < 8; ++r) ts[r + hi * 8][n0 + nl] = (_Float16)(acc[r] + bv);
  }
  __syncthreads();

  // stage 2: z(16x128) = ts @ Wproj + bproj  (8 N-subtiles, 2 per wave)
  for (int s = wave; s < PP / 16; s += 4) {
    int n0 = s * 16;
    v8f acc = {};
#pragma unroll
    for (int k0 = 0; k0 < ENC; k0 += 32) {
      v16h a = load_a_h(&ts[0][0], ENC, k0, lane);
      v16h b = load_b_t(WprojT, ENC, n0, k0, lane);
      acc = wmma_step(a, b, acc);
    }
    float bv = bproj[n0 + nl];
#pragma unroll
    for (int r = 0; r < 8; ++r)
      z[(size_t)(m0 + r + hi * 8) * PP + n0 + nl] = (_Float16)(acc[r] + bv);
  }
}

// ---------------------------------------------------------------------------
// Kernel 2: gi = z @ Wi + bi  (M=B*T, K=128, N=768).
// One wave per 16x64 strip: A-fragment reused across 4 B-tiles per K step.
// ---------------------------------------------------------------------------
__global__ void gi_kernel(const _Float16* __restrict__ z, const _Float16* __restrict__ WiT,
                          const float* __restrict__ bi, float* __restrict__ gi) {
  const int lane = threadIdx.x & 31;
  const int m0 = blockIdx.x * 16;
  const int nb = blockIdx.y * 64;
  __builtin_prefetch(WiT + (size_t)nb * PP, 0, 0);  // global_prefetch_b8

  v8f acc[4] = {{}, {}, {}, {}};
#pragma unroll
  for (int k0 = 0; k0 < PP; k0 += 32) {
    v16h a = load_a_h(z + (size_t)m0 * PP, PP, k0, lane);
#pragma unroll
    for (int s = 0; s < 4; ++s) {
      v16h b = load_b_t(WiT, PP, nb + s * 16, k0, lane);
      acc[s] = wmma_step(a, b, acc[s]);
    }
  }
  const int nl = lane & 15, hi = lane >> 4;
#pragma unroll
  for (int s = 0; s < 4; ++s) {
    int n0 = nb + s * 16;
    float bv = bi[n0 + nl];
#pragma unroll
    for (int r = 0; r < 8; ++r)
      gi[(size_t)(m0 + r + hi * 8) * (3 * HH) + n0 + nl] = acc[s][r] + bv;
  }
}

// ---------------------------------------------------------------------------
// Kernel 3: GRU scan. One workgroup (256 threads) per batch element.
// gh = h @ Wh uses f16 Wh^T with contiguous b128 loads; h kept fp32 in LDS.
// ---------------------------------------------------------------------------
__global__ void gru_kernel(const float* __restrict__ gi, const _Float16* __restrict__ WhT,
                           const float* __restrict__ bhn, _Float16* __restrict__ c) {
  __shared__ float h[HH];
  __shared__ float gh[3 * HH];
  const int b = blockIdx.x, tid = threadIdx.x;

  h[tid] = 0.0f;
  __syncthreads();

  const float bh = bhn[tid];
  for (int t = 0; t < TT; ++t) {
#pragma unroll
    for (int part = 0; part < 3; ++part) {
      int j = tid + part * HH;
      const _Float16* w = WhT + (size_t)j * HH;  // row of Wh^T, contiguous over i
      float acc = 0.0f;
      for (int i0 = 0; i0 < HH; i0 += 8) {
        h8 wv = *(const h8*)(w + i0);            // global_load_b128
#pragma unroll
        for (int u = 0; u < 8; ++u) acc = fmaf(h[i0 + u], (float)wv[u], acc);
      }
      gh[j] = acc;
    }
    __syncthreads();

    const float* g = gi + ((size_t)b * TT + t) * (3 * HH);
    float ir = g[tid], iz = g[tid + HH], inn = g[tid + 2 * HH];
    float r  = 1.0f / (1.0f + __expf(-(ir + gh[tid])));
    float zz = 1.0f / (1.0f + __expf(-(iz + gh[tid + HH])));
    float n  = tanhf(inn + r * (gh[tid + 2 * HH] + bh));
    float hn = (1.0f - zz) * n + zz * h[tid];

    h[tid] = hn;
    c[((size_t)b * TT + t) * HH + tid] = (_Float16)hn;
    __syncthreads();
  }
}

// ---------------------------------------------------------------------------
// Kernel 4: per (t-tile, b, k) loss partial.
//   pred(16x128) = c[b, t0:t0+16] @ Wp[k-1] + bp[k-1]   -> LDS f16
//   stream 16x16 logits tiles (pred @ z_pos^T)/TEMP with online per-row
//   max / sum-exp / sum;  partial = sum_rows(LSE - S/Tk) / (B*Tk*K)
// Grid (32, 64, 12), one wave per block.
// ---------------------------------------------------------------------------
__global__ void loss_kernel(const _Float16* __restrict__ c, const _Float16* __restrict__ z,
                            const _Float16* __restrict__ WpT, const float* __restrict__ bp,
                            float* __restrict__ partials) {
  __shared__ __align__(16) _Float16 predS[16][PP];   // 4 KB
  __shared__ float logS[16][16];
  __shared__ float red[16];

  const int lane = threadIdx.x & 31;
  const int nl = lane & 15, hi = lane >> 4;
  const int t0 = blockIdx.x * 16;
  const int b  = blockIdx.y;
  const int kk = blockIdx.z + 1;           // prediction horizon 1..12
  const int Tk = TT - kk;
  const int flat = (blockIdx.z * 64 + blockIdx.y) * 32 + blockIdx.x;

  // phase 1: pred tile into LDS (f16)
  const _Float16* Crow = c + ((size_t)b * TT + t0) * HH;
  const _Float16* WpkT = WpT + (size_t)(kk - 1) * PP * HH;  // [P][H] f16
  const float*    bpk  = bp + (size_t)(kk - 1) * PP;
  for (int n0 = 0; n0 < PP; n0 += 16) {
    v8f acc = {};
#pragma unroll
    for (int k0 = 0; k0 < HH; k0 += 32) {
      v16h a = load_a_h(Crow, HH, k0, lane);
      v16h bb = load_b_t(WpkT, HH, n0, k0, lane);
      acc = wmma_step(a, bb, acc);
    }
    float bv = bpk[n0 + nl];
#pragma unroll
    for (int r = 0; r < 8; ++r) predS[r + hi * 8][n0 + nl] = (_Float16)(acc[r] + bv);
  }
  __syncthreads();

  // phase 2: logits tiles with online LSE / row-sum (z f16 == z_pos^T form)
  const _Float16* Zb = z + ((size_t)b * TT + kk) * PP;
  const int maxrow = Tk - 1;                    // clamp keeps reads in-bounds
  float mx = -INFINITY, se = 0.0f, srow = 0.0f; // per-lane row state (lanes 0..15)

  for (int jt = 0; jt < 32; ++jt) {
    int n0 = jt * 16;
    v8f acc = {};
#pragma unroll
    for (int k0 = 0; k0 < PP; k0 += 32) {
      v16h a = load_a_h(&predS[0][0], PP, k0, lane);
      v16h bb = load_b_t_clamp(Zb, PP, n0, k0, maxrow, lane);
      acc = wmma_step(a, bb, acc);
    }
#pragma unroll
    for (int r = 0; r < 8; ++r) logS[r + hi * 8][nl] = acc[r] * INV_TEMP;
    __syncthreads();

    if (lane < 16) {
#pragma unroll
      for (int cc = 0; cc < 16; ++cc) {
        int jc = n0 + cc;
        if (jc < Tk) {
          float v = logS[lane][cc];
          if (v > mx) { se *= __expf(mx - v); mx = v; }
          se += __expf(v - mx);
          srow += v;
        }
      }
    }
    __syncthreads();
  }

  float contrib = 0.0f;
  if (lane < 16 && (t0 + lane) < Tk)
    contrib = (mx + __logf(se)) - srow / (float)Tk;
  if (lane < 16) red[lane] = contrib;
  __syncthreads();

  if (lane == 0) {
    float s = 0.0f;
#pragma unroll
    for (int i = 0; i < 16; ++i) s += red[i];
    partials[flat] = s / ((float)BB * (float)Tk * (float)KP);
  }
}

// ---------------------------------------------------------------------------
// Kernel 5: deterministic fixed-order reduction of 24576 partials -> d_out[0]
// ---------------------------------------------------------------------------
__global__ void reduce_kernel(const float* __restrict__ partials, float* __restrict__ out) {
  __shared__ float red[256];
  const int tid = threadIdx.x;
  const int N = KP * 64 * 32;
  float s = 0.0f;
  for (int i = tid; i < N; i += 256) s += partials[i];
  red[tid] = s;
  __syncthreads();
  if (tid == 0) {
    float tot = 0.0f;
    for (int i = 0; i < 256; ++i) tot += red[i];
    out[0] = tot;
  }
}

// ---------------------------------------------------------------------------
extern "C" void kernel_launch(void* const* d_in, const int* in_sizes, int n_in,
                              void* d_out, int out_size, void* d_ws, size_t ws_size,
                              hipStream_t stream) {
  const float* x     = (const float*)d_in[0];   // (B,T,F)
  const float* Wenc  = (const float*)d_in[1];   // (F,ENC)
  const float* benc  = (const float*)d_in[2];   // (ENC,)
  const float* Wproj = (const float*)d_in[3];   // (ENC,P)
  const float* bproj = (const float*)d_in[4];   // (P,)
  const float* Wi    = (const float*)d_in[5];   // (P,3H)
  const float* bi    = (const float*)d_in[6];   // (3H,)
  const float* Wh    = (const float*)d_in[7];   // (H,3H)
  const float* bhn   = (const float*)d_in[8];   // (H,)
  const float* Wp    = (const float*)d_in[9];   // (K,H,P)
  const float* bp    = (const float*)d_in[10];  // (K,P)
  float* out = (float*)d_out;

  // workspace carve (all regions 16B aligned)
  float* gi       = (float*)d_ws;                         // B*T*3H f32 = 96 MB
  float* partials = gi + (size_t)BB * TT * 3 * HH;        // 24576 f32
  _Float16* z     = (_Float16*)(partials + KP * 64 * 32); // B*T*P  f16
  _Float16* c     = z  + (size_t)BB * TT * PP;            // B*T*H  f16
  _Float16* xh    = c  + (size_t)BB * TT * HH;            // B*T*F  f16
  _Float16* WencT = xh + (size_t)BB * TT * FF;            // ENCxF
  _Float16* WprojT= WencT + (size_t)ENC * FF;             // PxENC
  _Float16* WiT   = WprojT + (size_t)PP * ENC;            // 3HxP
  _Float16* WhT   = WiT + (size_t)3 * HH * PP;            // 3HxH
  _Float16* WpT   = WhT + (size_t)3 * HH * HH;            // K x P x H

  const int Mtiles = (BB * TT) / 16;  // 2048

  // one-time f16 conversion / transposition of operands
  cvt_kernel<<<4096, 256, 0, stream>>>(x, xh, BB * TT * FF);
  tr_kernel<<<dim3((FF * ENC + 255) / 256, 1), 256, 0, stream>>>(Wenc, WencT, FF, ENC, 0, 0);
  tr_kernel<<<dim3((ENC * PP + 255) / 256, 1), 256, 0, stream>>>(Wproj, WprojT, ENC, PP, 0, 0);
  tr_kernel<<<dim3((PP * 3 * HH + 255) / 256, 1), 256, 0, stream>>>(Wi, WiT, PP, 3 * HH, 0, 0);
  tr_kernel<<<dim3((HH * 3 * HH + 255) / 256, 1), 256, 0, stream>>>(Wh, WhT, HH, 3 * HH, 0, 0);
  tr_kernel<<<dim3((HH * PP + 255) / 256, KP), 256, 0, stream>>>(Wp, WpT, HH, PP, HH * PP, HH * PP);

  enc_kernel<<<Mtiles, 128, 0, stream>>>(xh, WencT, benc, WprojT, bproj, z);
  gi_kernel<<<dim3(Mtiles, (3 * HH) / 64), 32, 0, stream>>>(z, WiT, bi, gi);
  gru_kernel<<<BB, HH, 0, stream>>>(gi, WhT, bhn, c);
  loss_kernel<<<dim3(32, BB, KP), 32, 0, stream>>>(c, z, WpT, bp, partials);
  reduce_kernel<<<1, 256, 0, stream>>>(partials, out);
}
